// HybridGNNAnomalyDetector_19086834663832
// MI455X (gfx1250) — matrix operand order repaired and checked by reading, not compiled, not run on hardware
//
#include <hip/hip_runtime.h>
#include <hip/hip_bf16.h>
#include <math.h>

typedef __attribute__((ext_vector_type(16))) _Float16 v16h;
typedef __attribute__((ext_vector_type(2)))  __fp16   v2fp;  // cvt_pkrtz return type
typedef __attribute__((ext_vector_type(8)))  float    v8f;

// ---------------------------------------------------------------- utilities

__device__ __forceinline__ void atomicMaxF(float* addr, float v){
  if (v >= 0.0f) atomicMax((int*)addr, __float_as_int(v));
  else           atomicMin((unsigned int*)addr, __float_as_uint(v));
}

__device__ __forceinline__ unsigned pkh2(float a, float b){
  // single v_cvt_pk_rtz_f16_f32 -- no scalar-half merge ops
  union { v2fp h; unsigned u; } x;
  x.h = __builtin_amdgcn_cvt_pkrtz(a, b);
  return x.u;
}

union FragU {          // 16 halves viewed as 2x uint4 for b128 LDS traffic
  v16h h;
  uint4 q[2];
};

__global__ void k_fill(float* __restrict__ p, float v, long long n){
  long long i = (long long)blockIdx.x * blockDim.x + threadIdx.x;
  if (i < n) p[i] = v;
}

__global__ void k_addbias(float* __restrict__ p, const float* __restrict__ b,
                          long long n, int cols){
  long long i = (long long)blockIdx.x * blockDim.x + threadIdx.x;
  if (i < n) p[i] += b[(int)(i % cols)];
}

// ------------------------------------------------- edge_attr column sums (16)

__global__ void k_colsum16(const float* __restrict__ eattr, float* __restrict__ esum,
                           int E, int rowsPerBlock){
  __shared__ float sh[16][16];
  int t = threadIdx.x;            // 256 threads
  int col = t & 15, rg = t >> 4;
  int r0 = blockIdx.x * rowsPerBlock;
  float s = 0.0f;
  for (int i = rg; i < rowsPerBlock; i += 16){
    int r = r0 + i;
    if (r < E) s += eattr[(size_t)r * 16 + col];
  }
  sh[rg][col] = s;
  __syncthreads();
  if (t < 16){
    float a = 0.0f;
    #pragma unroll
    for (int j = 0; j < 16; ++j) a += sh[j][t];
    atomicAdd(&esum[t], a);
  }
}

// ------------------------- fold edge-attention: M1[16,4], m2[16], self consts

__global__ void k_precompute(const float* __restrict__ esum,
                             const float* __restrict__ We1, const float* __restrict__ ae1,
                             const float* __restrict__ We2, const float* __restrict__ ae2,
                             float* __restrict__ emean, float* __restrict__ M1,
                             float* __restrict__ m2, float* __restrict__ selfc, int E){
  int t = threadIdx.x; // 64 threads
  if (t < 16) emean[t] = esum[t] / (float)E;
  {
    int d = t >> 2, h = t & 3;
    float s = 0.0f;
    for (int o = 0; o < 64; ++o) s += We1[(size_t)d * 256 + h * 64 + o] * ae1[h * 64 + o];
    M1[t] = s;
  }
  if (t < 16){
    float s = 0.0f;
    for (int o = 0; o < 32; ++o) s += We2[(size_t)t * 32 + o] * ae2[o];
    m2[t] = s;
  }
  __syncthreads();
  if (t < 4){
    float s = 0.0f;
    for (int d = 0; d < 16; ++d) s += emean[d] * M1[d * 4 + t];
    selfc[t] = s;
  }
  if (t == 0){
    float s = 0.0f;
    for (int d = 0; d < 16; ++d) s += emean[d] * m2[d];
    selfc[4] = s;
  }
}

// ----------------------------------------------------- WMMA f16->f32 GEMM
// C[M,N] = act(A[M,K] @ B[K,N] + bias). Block = 8 waves; block tile 128x16,
// K stepped by 64 (two v_wmma_f32_16x16x32_f16 per barrier pair).
// LDS holds f16 PAIRS (uint): Asu[row][k/2], Bsu[col][k/2]; stride 36 uints
// (144B) keeps uint4 alignment and spreads rows across banks.
// Fragment fetch = 2x ds_load_b128 per operand (ISA 7.12.2 pair runs).

#define KSTEP 64
#define LDU   36   // padded LDS row stride in uints

__device__ __forceinline__ void ld_frags_and_mma(
    const unsigned (*Asu)[LDU], const unsigned (*Bsu)[LDU],
    int mloc, int nloc, int half, v8f& acc){
  #pragma unroll
  for (int s = 0; s < 2; ++s){
    FragU a_, b_;
    a_.q[0] = *(const uint4*)&Asu[mloc][s * 16 + half * 4];
    a_.q[1] = *(const uint4*)&Asu[mloc][s * 16 + 8 + half * 4];
    b_.q[0] = *(const uint4*)&Bsu[nloc][s * 16 + half * 4];
    b_.q[1] = *(const uint4*)&Bsu[nloc][s * 16 + 8 + half * 4];
    acc = __builtin_amdgcn_wmma_f32_16x16x32_f16(false, a_.h, false, b_.h,
                                                 (short)0, acc, false, false);
  }
}

__device__ __forceinline__ void store_a16(unsigned* dst, const float* src, bool inb){
  // one 16-wide K group -> 8 packed uints as 2x uint4 (global b128 loads)
  uint4 o0, o1;
  if (inb){
    const float4* s4 = (const float4*)src;
    float4 f0 = s4[0], f1 = s4[1], f2 = s4[2], f3 = s4[3];
    o0.x = pkh2(f0.x, f0.y); o0.y = pkh2(f0.z, f0.w);
    o0.z = pkh2(f1.x, f1.y); o0.w = pkh2(f1.z, f1.w);
    o1.x = pkh2(f2.x, f2.y); o1.y = pkh2(f2.z, f2.w);
    o1.z = pkh2(f3.x, f3.y); o1.w = pkh2(f3.z, f3.w);
  } else {
    o0.x = o0.y = o0.z = o0.w = 0u;
    o1 = o0;
  }
  ((uint4*)dst)[0] = o0;
  ((uint4*)dst)[1] = o1;
}

template<int EPI> // 0=none 1=bias 2=bias+relu
__global__ __launch_bounds__(256)
void k_gemm(const float* __restrict__ A, const float* __restrict__ B,
            const float* __restrict__ bias, float* __restrict__ C,
            int M, int N, int K){
  __shared__ unsigned Asu[128][LDU];
  __shared__ unsigned Bsu[16][LDU];
  int tid = threadIdx.x;
  int wave = tid >> 5, lane = tid & 31;
  int gm0 = blockIdx.x * 128;
  int gn0 = blockIdx.y * 16;
  v8f acc = {};
  int ksteps = (K + KSTEP - 1) / KSTEP;
  int ar = tid >> 1;                 // A row handled by this thread
  int acb = (tid & 1) * 16;          // first 16-col group
  int mloc = wave * 16 + (lane & 15);
  int nloc = lane & 15;
  int half = lane >> 4;
  for (int ks = 0; ks < ksteps; ++ks){
    int k0 = ks * KSTEP;
    { // A tile 128x64 halves: 2 groups of 16 cols per thread
      int gr = gm0 + ar;
      const float* Ar = A + (size_t)gr * K;
      #pragma unroll
      for (int g = 0; g < 2; ++g){
        int cb = acb + g * 32;
        int kg = k0 + cb;
        bool inb = (gr < M) && (kg < K);   // K is a multiple of 16 -> whole group
        store_a16(&Asu[ar][cb >> 1], Ar + kg, inb);
      }
    }
    { // B tile 64x16 -> packed pairs per column; 2 uints per thread
      #pragma unroll
      for (int q = 0; q < 2; ++q){
        int idx = tid * 2 + q;          // 0..511
        int c = idx & 15, ku = idx >> 4; // pair index 0..31
        int gk = k0 + ku * 2, gc = gn0 + c;
        float v0 = 0.0f, v1 = 0.0f;
        if (gc < N){
          if (gk < K)     v0 = B[(size_t)gk * N + gc];
          if (gk + 1 < K) v1 = B[(size_t)(gk + 1) * N + gc];
        }
        Bsu[c][ku] = pkh2(v0, v1);
      }
    }
    __syncthreads();
    ld_frags_and_mma(Asu, Bsu, mloc, nloc, half, acc);
    __syncthreads();
  }
  int gc = gn0 + nloc;
  float bv = 0.0f;
  if (EPI >= 1 && gc < N) bv = bias[gc];
  #pragma unroll
  for (int r = 0; r < 8; ++r){
    int grow = gm0 + wave * 16 + r + (half << 3);
    if (grow < M && gc < N){
      float v = acc[r];
      if (EPI >= 1) v += bv;
      if (EPI == 2) v = v > 0.0f ? v : 0.0f;
      C[(size_t)grow * N + gc] = v;
    }
  }
}

// Gather-GEMM for edge decoder layer 1: logical A row e =
// [z[src[e]](32) | z[dst[e]](32) | edge_attr[e](16) | 0-pad]; K passed as 80.
template<int EPI>
__global__ __launch_bounds__(256)
void k_gemm_gather(const float* __restrict__ z, const int* __restrict__ srcI,
                   const int* __restrict__ dstI, const float* __restrict__ eattr,
                   const float* __restrict__ B, const float* __restrict__ bias,
                   float* __restrict__ C, int chunkBase, int M, int N, int K){
  __shared__ unsigned Asu[128][LDU];
  __shared__ unsigned Bsu[16][LDU];
  int tid = threadIdx.x;
  int wave = tid >> 5, lane = tid & 31;
  int gm0 = blockIdx.x * 128;
  int gn0 = blockIdx.y * 16;
  v8f acc = {};
  int ksteps = (K + KSTEP - 1) / KSTEP;
  int ar = tid >> 1;
  int acb = (tid & 1) * 16;
  int mloc = wave * 16 + (lane & 15);
  int nloc = lane & 15;
  int half = lane >> 4;
  int gr = gm0 + ar;
  int e = chunkBase + gr;
  for (int ks = 0; ks < ksteps; ++ks){
    int k0 = ks * KSTEP;
    {
      #pragma unroll
      for (int g = 0; g < 2; ++g){
        int cb = acb + g * 32;
        int kg = k0 + cb;               // 16-wide group: single source each
        const float* sp = nullptr;
        if (gr < M){
          if      (kg < 32) sp = z + (size_t)srcI[e] * 32 + kg;
          else if (kg < 64) sp = z + (size_t)dstI[e] * 32 + (kg - 32);
          else if (kg < 80) sp = eattr + (size_t)e * 16 + (kg - 64);
        }
        store_a16(&Asu[ar][cb >> 1], sp, sp != nullptr);
      }
    }
    {
      #pragma unroll
      for (int q = 0; q < 2; ++q){
        int idx = tid * 2 + q;
        int c = idx & 15, ku = idx >> 4;
        int gk = k0 + ku * 2, gc = gn0 + c;
        float v0 = 0.0f, v1 = 0.0f;
        if (gc < N){
          if (gk < K)     v0 = B[(size_t)gk * N + gc];
          if (gk + 1 < K) v1 = B[(size_t)(gk + 1) * N + gc];
        }
        Bsu[c][ku] = pkh2(v0, v1);
      }
    }
    __syncthreads();
    ld_frags_and_mma(Asu, Bsu, mloc, nloc, half, acc);
    __syncthreads();
  }
  int gc = gn0 + nloc;
  float bv = 0.0f;
  if (EPI >= 1 && gc < N) bv = bias[gc];
  #pragma unroll
  for (int r = 0; r < 8; ++r){
    int grow = gm0 + wave * 16 + r + (half << 3);
    if (grow < M && gc < N){
      float v = acc[r];
      if (EPI >= 1) v += bv;
      if (EPI == 2) v = v > 0.0f ? v : 0.0f;
      C[(size_t)grow * N + gc] = v;
    }
  }
}

// ------------------------------------------- per-node attention coefficients

__global__ void k_rowdot4(const float* __restrict__ xs, const float* __restrict__ as_,
                          const float* __restrict__ ad_, float* __restrict__ osrc,
                          float* __restrict__ odst, int Nn){
  int w = (blockIdx.x * blockDim.x + threadIdx.x) >> 5;
  int l = threadIdx.x & 31;
  if (w >= Nn) return;
  const float* xr = xs + (size_t)w * 256 + l * 8;
  float ps = 0.0f, pd = 0.0f;
  #pragma unroll
  for (int j = 0; j < 8; ++j){
    float v = xr[j];
    ps += v * as_[l * 8 + j];
    pd += v * ad_[l * 8 + j];
  }
  for (int o = 1; o < 8; o <<= 1){ ps += __shfl_xor(ps, o, 32); pd += __shfl_xor(pd, o, 32); }
  if ((l & 7) == 0){ osrc[w * 4 + (l >> 3)] = ps; odst[w * 4 + (l >> 3)] = pd; }
}

__global__ void k_rowdot1(const float* __restrict__ xs, const float* __restrict__ as_,
                          const float* __restrict__ ad_, float* __restrict__ osrc,
                          float* __restrict__ odst, int Nn){
  int w = (blockIdx.x * blockDim.x + threadIdx.x) >> 5;
  int l = threadIdx.x & 31;
  if (w >= Nn) return;
  float v = xs[(size_t)w * 32 + l];
  float ps = v * as_[l], pd = v * ad_[l];
  for (int o = 1; o < 32; o <<= 1){ ps += __shfl_xor(ps, o, 32); pd += __shfl_xor(pd, o, 32); }
  if (l == 0){ osrc[w] = ps; odst[w] = pd; }
}

// ---------------------------------------- attention: alpha + segment max/sum

__global__ void k_alpha4(const int* __restrict__ srcI, const int* __restrict__ dstI,
                         const float* __restrict__ eattr, const float* __restrict__ M1,
                         const float* __restrict__ selfc, const float* __restrict__ a_src,
                         const float* __restrict__ a_dst, float* __restrict__ alpha,
                         float* __restrict__ mx, int E, int Nn){
  __shared__ float sM[64];
  int t = threadIdx.x;
  if (t < 64) sM[t] = M1[t];
  __syncthreads();
  int e = blockIdx.x * blockDim.x + t;
  if (e >= E + Nn) return;
  int s, d; float ae0, ae1, ae2, ae3;
  if (e < E){
    s = srcI[e]; d = dstI[e];
    ae0 = ae1 = ae2 = ae3 = 0.0f;
    const float* er = eattr + (size_t)e * 16;
    #pragma unroll
    for (int dd = 0; dd < 16; ++dd){
      float ev = er[dd];
      ae0 += ev * sM[dd * 4 + 0]; ae1 += ev * sM[dd * 4 + 1];
      ae2 += ev * sM[dd * 4 + 2]; ae3 += ev * sM[dd * 4 + 3];
    }
  } else {
    s = d = e - E;
    ae0 = selfc[0]; ae1 = selfc[1]; ae2 = selfc[2]; ae3 = selfc[3];
  }
  float ae[4] = {ae0, ae1, ae2, ae3};
  #pragma unroll
  for (int h = 0; h < 4; ++h){
    float a = a_src[s * 4 + h] + a_dst[d * 4 + h] + ae[h];
    a = a > 0.0f ? a : 0.2f * a;
    alpha[(size_t)e * 4 + h] = a;
    atomicMaxF(&mx[d * 4 + h], a);
  }
}

__global__ void k_alpha1(const int* __restrict__ srcI, const int* __restrict__ dstI,
                         const float* __restrict__ eattr, const float* __restrict__ m2,
                         const float* __restrict__ selfc, const float* __restrict__ a_src,
                         const float* __restrict__ a_dst, float* __restrict__ alpha,
                         float* __restrict__ mx, int E, int Nn){
  __shared__ float sm[16];
  int t = threadIdx.x;
  if (t < 16) sm[t] = m2[t];
  __syncthreads();
  int e = blockIdx.x * blockDim.x + t;
  if (e >= E + Nn) return;
  int s, d; float ae;
  if (e < E){
    s = srcI[e]; d = dstI[e];
    ae = 0.0f;
    const float* er = eattr + (size_t)e * 16;
    #pragma unroll
    for (int dd = 0; dd < 16; ++dd) ae += er[dd] * sm[dd];
  } else { s = d = e - E; ae = selfc[4]; }
  float a = a_src[s] + a_dst[d] + ae;
  a = a > 0.0f ? a : 0.2f * a;
  alpha[e] = a;
  atomicMaxF(&mx[d], a);
}

__global__ void k_expsum(const int* __restrict__ dstI, float* __restrict__ p,
                         const float* __restrict__ mx, float* __restrict__ den,
                         int E, int Nn, int H){
  int e = blockIdx.x * blockDim.x + threadIdx.x;
  if (e >= E + Nn) return;
  int d = (e < E) ? dstI[e] : e - E;
  for (int h = 0; h < H; ++h){
    float v = expf(p[(size_t)e * H + h] - mx[d * H + h]);
    p[(size_t)e * H + h] = v;
    atomicAdd(&den[d * H + h], v);
  }
}

// ------------------------------------------------ weighted scatter-aggregate

__global__ void k_agg4(const int* __restrict__ srcI, const int* __restrict__ dstI,
                       const float* __restrict__ p, const float* __restrict__ den,
                       const float* __restrict__ xs, float* __restrict__ agg,
                       int E, int Nn){
  int wid = (blockIdx.x * blockDim.x + threadIdx.x) >> 5;
  int l = threadIdx.x & 31;
  if (wid >= E + Nn) return;
  int s, d;
  if (wid < E){ s = srcI[wid]; d = dstI[wid]; } else { s = d = wid - E; }
  int h = l >> 3;
  float w = p[(size_t)wid * 4 + h] / den[d * 4 + h];
  const float* xr = xs + (size_t)s * 256 + l * 8;
  float* orow = agg + (size_t)d * 256 + l * 8;
  #pragma unroll
  for (int j = 0; j < 8; ++j) atomicAdd(&orow[j], w * xr[j]);
}

__global__ void k_agg1(const int* __restrict__ srcI, const int* __restrict__ dstI,
                       const float* __restrict__ p, const float* __restrict__ den,
                       const float* __restrict__ xs, float* __restrict__ agg,
                       int E, int Nn){
  int wid = (blockIdx.x * blockDim.x + threadIdx.x) >> 5;
  int l = threadIdx.x & 31;
  if (wid >= E + Nn) return;
  int s, d;
  if (wid < E){ s = srcI[wid]; d = dstI[wid]; } else { s = d = wid - E; }
  float w = p[wid] / den[d];
  atomicAdd(&agg[(size_t)d * 32 + l], w * xs[(size_t)s * 32 + l]);
}

// ------------------------------------------------------------- batch norm

__global__ void k_bnstats(const float* __restrict__ X, float* __restrict__ sum,
                          float* __restrict__ sq, int rows, int cols, int rowsPerBlock){
  int c = threadIdx.x; // blockDim == cols
  int r0 = blockIdx.x * rowsPerBlock;
  float s = 0.0f, s2 = 0.0f;
  for (int i = 0; i < rowsPerBlock; ++i){
    int r = r0 + i;
    if (r < rows){ float v = X[(size_t)r * cols + c]; s += v; s2 += v * v; }
  }
  atomicAdd(&sum[c], s);
  atomicAdd(&sq[c], s2);
}

__global__ void k_bnfin(const float* __restrict__ sum, const float* __restrict__ sq,
                        const float* __restrict__ g, const float* __restrict__ b,
                        float* __restrict__ scale, float* __restrict__ shift,
                        float cnt, int cols){
  int c = threadIdx.x;
  if (c >= cols) return;
  float m = sum[c] / cnt;
  float v = sq[c] / cnt - m * m;
  float sc = g[c] * rsqrtf(v + 1e-5f);
  scale[c] = sc;
  shift[c] = b[c] - m * sc;
}

template<int MODE> // 0 = plain, 1 = ELU after
__global__ void k_bnapply(float* __restrict__ X, const float* __restrict__ scale,
                          const float* __restrict__ shift, long long n, int cols){
  long long i = (long long)blockIdx.x * blockDim.x + threadIdx.x;
  if (i >= n) return;
  int c = (int)(i % cols);
  float y = X[i] * scale[c] + shift[c];
  if (MODE == 1) y = y > 0.0f ? y : (expf(y) - 1.0f);
  X[i] = y;
}

// -------------------------------------------- final edge layer: dot64+sigmoid

__global__ void k_dot64sig(const float* __restrict__ H2, const float* __restrict__ w3,
                           const float* __restrict__ b3, float* __restrict__ out, int M){
  int r = (blockIdx.x * blockDim.x + threadIdx.x) >> 5;
  int l = threadIdx.x & 31;
  if (r >= M) return;
  const float* hr = H2 + (size_t)r * 64;
  float p = hr[l] * w3[l] + hr[32 + l] * w3[32 + l];
  for (int o = 1; o < 32; o <<= 1) p += __shfl_xor(p, o, 32);
  if (l == 0) out[r] = 1.0f / (1.0f + expf(-(p + b3[0])));
}

// ================================================================== launcher

extern "C" void kernel_launch(void* const* d_in, const int* in_sizes, int n_in,
                              void* d_out, int out_size, void* d_ws, size_t ws_size,
                              hipStream_t stream){
  const float* x      = (const float*)d_in[0];
  const int*   eidx   = (const int*)  d_in[1];
  const float* eattr  = (const float*)d_in[2];
  const float* W1     = (const float*)d_in[3];
  const float* att_s1 = (const float*)d_in[4];
  const float* att_d1 = (const float*)d_in[5];
  const float* We1    = (const float*)d_in[6];
  const float* att_e1 = (const float*)d_in[7];
  const float* b1     = (const float*)d_in[8];
  const float* bn1_g  = (const float*)d_in[9];
  const float* bn1_b  = (const float*)d_in[10];
  const float* W2     = (const float*)d_in[11];
  const float* att_s2 = (const float*)d_in[12];
  const float* att_d2 = (const float*)d_in[13];
  const float* We2    = (const float*)d_in[14];
  const float* att_e2 = (const float*)d_in[15];
  const float* b2     = (const float*)d_in[16];
  const float* nd_w1  = (const float*)d_in[17];
  const float* nd_b1  = (const float*)d_in[18];
  const float* nd_g   = (const float*)d_in[19];
  const float* nd_bb  = (const float*)d_in[20];
  const float* nd_w2  = (const float*)d_in[21];
  const float* nd_b2  = (const float*)d_in[22];
  const float* nd_w3  = (const float*)d_in[23];
  const float* nd_b3  = (const float*)d_in[24];
  const float* ed_w1  = (const float*)d_in[25];
  const float* ed_b1  = (const float*)d_in[26];
  const float* ed_g   = (const float*)d_in[27];
  const float* ed_bb  = (const float*)d_in[28];
  const float* ed_w2  = (const float*)d_in[29];
  const float* ed_b2  = (const float*)d_in[30];
  const float* ed_w3  = (const float*)d_in[31];
  const float* ed_b3  = (const float*)d_in[32];

  const int Nn = in_sizes[0] / 128;
  const int E  = in_sizes[1] / 2;
  const int E2 = E + Nn;
  const int* srcI = eidx;
  const int* dstI = eidx + E;

  float* Wp = (float*)d_ws;
  size_t off = 0;
  auto alloc = [&](size_t n)->float*{ float* p = Wp + off; off += n; return p; };
  float* esum   = alloc(16);
  float* emean  = alloc(16);
  float* M1     = alloc(64);
  float* m2     = alloc(16);
  float* selfc  = alloc(8);
  float* bn_sum = alloc(256);
  float* bn_sq  = alloc(256);
  float* bn_sc  = alloc(256);
  float* bn_sh  = alloc(256);
  float* a_src1 = alloc((size_t)Nn * 4);
  float* a_dst1 = alloc((size_t)Nn * 4);
  float* mx1    = alloc((size_t)Nn * 4);
  float* den1   = alloc((size_t)Nn * 4);
  float* p1     = alloc((size_t)E2 * 4);
  float* a_src2 = alloc(Nn);
  float* a_dst2 = alloc(Nn);
  float* mx2    = alloc(Nn);
  float* den2   = alloc(Nn);
  float* p2     = alloc(E2);
  float* xs1    = alloc((size_t)Nn * 256);
  float* h1b    = alloc((size_t)Nn * 256);
  float* xs2    = alloc((size_t)Nn * 32);
  float* ndh1   = alloc((size_t)Nn * 128);
  float* ndh2   = alloc((size_t)Nn * 64);
  const int CHUNK = 64000;
  float* edh1   = alloc((size_t)CHUNK * 128);
  float* edh2   = alloc((size_t)CHUNK * 64);
  (void)ws_size; (void)n_in; (void)out_size;

  float* zOut  = (float*)d_out;
  float* nrOut = zOut + (size_t)Nn * 32;
  float* erOut = nrOut + (size_t)Nn * 128;

  auto blks = [](long long n, int b){ return (unsigned)((n + b - 1) / b); };
  auto g2   = [](int M, int Ncol){ return dim3((unsigned)((M + 127) / 128),
                                               (unsigned)((Ncol + 15) / 16)); };

  // ---- folded edge-attention constants
  k_fill<<<1, 256, 0, stream>>>(esum, 0.0f, 16);
  k_colsum16<<<blks(E, 4096), 256, 0, stream>>>(eattr, esum, E, 4096);
  k_precompute<<<1, 64, 0, stream>>>(esum, We1, att_e1, We2, att_e2,
                                     emean, M1, m2, selfc, E);

  // ---- GAT layer 1
  k_gemm<0><<<g2(Nn, 256), 256, 0, stream>>>(x, W1, nullptr, xs1, Nn, 256, 128);
  k_rowdot4<<<blks(Nn, 8), 256, 0, stream>>>(xs1, att_s1, att_d1, a_src1, a_dst1, Nn);
  k_fill<<<blks((long long)Nn * 4, 256), 256, 0, stream>>>(mx1, -3.0e38f, (long long)Nn * 4);
  k_fill<<<blks((long long)Nn * 4, 256), 256, 0, stream>>>(den1, 0.0f, (long long)Nn * 4);
  k_fill<<<blks((long long)Nn * 256, 256), 256, 0, stream>>>(h1b, 0.0f, (long long)Nn * 256);
  k_alpha4<<<blks(E2, 256), 256, 0, stream>>>(srcI, dstI, eattr, M1, selfc,
                                              a_src1, a_dst1, p1, mx1, E, Nn);
  k_expsum<<<blks(E2, 256), 256, 0, stream>>>(dstI, p1, mx1, den1, E, Nn, 4);
  k_agg4<<<blks(E2, 8), 256, 0, stream>>>(srcI, dstI, p1, den1, xs1, h1b, E, Nn);
  k_addbias<<<blks((long long)Nn * 256, 256), 256, 0, stream>>>(h1b, b1, (long long)Nn * 256, 256);
  k_fill<<<1, 256, 0, stream>>>(bn_sum, 0.0f, 256);
  k_fill<<<1, 256, 0, stream>>>(bn_sq, 0.0f, 256);
  k_bnstats<<<blks(Nn, 256), 256, 0, stream>>>(h1b, bn_sum, bn_sq, Nn, 256, 256);
  k_bnfin<<<1, 256, 0, stream>>>(bn_sum, bn_sq, bn1_g, bn1_b, bn_sc, bn_sh, (float)Nn, 256);
  k_bnapply<1><<<blks((long long)Nn * 256, 256), 256, 0, stream>>>(h1b, bn_sc, bn_sh,
                                                                   (long long)Nn * 256, 256);

  // ---- GAT layer 2 -> z (written into d_out)
  k_gemm<0><<<g2(Nn, 32), 256, 0, stream>>>(h1b, W2, nullptr, xs2, Nn, 32, 256);
  k_rowdot1<<<blks(Nn, 8), 256, 0, stream>>>(xs2, att_s2, att_d2, a_src2, a_dst2, Nn);
  k_fill<<<blks(Nn, 256), 256, 0, stream>>>(mx2, -3.0e38f, Nn);
  k_fill<<<blks(Nn, 256), 256, 0, stream>>>(den2, 0.0f, Nn);
  k_fill<<<blks((long long)Nn * 32, 256), 256, 0, stream>>>(zOut, 0.0f, (long long)Nn * 32);
  k_alpha1<<<blks(E2, 256), 256, 0, stream>>>(srcI, dstI, eattr, m2, selfc,
                                              a_src2, a_dst2, p2, mx2, E, Nn);
  k_expsum<<<blks(E2, 256), 256, 0, stream>>>(dstI, p2, mx2, den2, E, Nn, 1);
  k_agg1<<<blks(E2, 8), 256, 0, stream>>>(srcI, dstI, p2, den2, xs2, zOut, E, Nn);
  k_addbias<<<blks((long long)Nn * 32, 256), 256, 0, stream>>>(zOut, b2, (long long)Nn * 32, 32);

  // ---- node decoder -> node_recon
  k_gemm<2><<<g2(Nn, 128), 256, 0, stream>>>(zOut, nd_w1, nd_b1, ndh1, Nn, 128, 32);
  k_fill<<<1, 256, 0, stream>>>(bn_sum, 0.0f, 128);
  k_fill<<<1, 256, 0, stream>>>(bn_sq, 0.0f, 128);
  k_bnstats<<<blks(Nn, 256), 128, 0, stream>>>(ndh1, bn_sum, bn_sq, Nn, 128, 256);
  k_bnfin<<<1, 256, 0, stream>>>(bn_sum, bn_sq, nd_g, nd_bb, bn_sc, bn_sh, (float)Nn, 128);
  k_bnapply<0><<<blks((long long)Nn * 128, 256), 256, 0, stream>>>(ndh1, bn_sc, bn_sh,
                                                                   (long long)Nn * 128, 128);
  k_gemm<2><<<g2(Nn, 64), 256, 0, stream>>>(ndh1, nd_w2, nd_b2, ndh2, Nn, 64, 128);
  k_gemm<1><<<g2(Nn, 128), 256, 0, stream>>>(ndh2, nd_w3, nd_b3, nrOut, Nn, 128, 64);

  // ---- edge decoder -> edge_recon (chunked; BN stats over full E)
  k_fill<<<1, 256, 0, stream>>>(bn_sum, 0.0f, 128);
  k_fill<<<1, 256, 0, stream>>>(bn_sq, 0.0f, 128);
  for (int c0 = 0; c0 < E; c0 += CHUNK){
    int mc = (E - c0) < CHUNK ? (E - c0) : CHUNK;
    k_gemm_gather<2><<<g2(mc, 128), 256, 0, stream>>>(zOut, srcI, dstI, eattr,
                                                      ed_w1, ed_b1, edh1, c0, mc, 128, 80);
    k_bnstats<<<blks(mc, 256), 128, 0, stream>>>(edh1, bn_sum, bn_sq, mc, 128, 256);
  }
  k_bnfin<<<1, 256, 0, stream>>>(bn_sum, bn_sq, ed_g, ed_bb, bn_sc, bn_sh, (float)E, 128);
  for (int c0 = 0; c0 < E; c0 += CHUNK){
    int mc = (E - c0) < CHUNK ? (E - c0) : CHUNK;
    k_gemm_gather<2><<<g2(mc, 128), 256, 0, stream>>>(zOut, srcI, dstI, eattr,
                                                      ed_w1, ed_b1, edh1, c0, mc, 128, 80);
    k_bnapply<0><<<blks((long long)mc * 128, 256), 256, 0, stream>>>(edh1, bn_sc, bn_sh,
                                                                     (long long)mc * 128, 128);
    k_gemm<2><<<g2(mc, 64), 256, 0, stream>>>(edh1, ed_w2, ed_b2, edh2, mc, 64, 128);
    k_dot64sig<<<blks(mc, 8), 256, 0, stream>>>(edh2, ed_w3, ed_b3, erOut + c0, mc);
  }
}
